// DecoderRNN_10539849744982
// MI455X (gfx1250) — matrix operand order repaired and checked by reading, not compiled, not run on hardware
//
#include <hip/hip_runtime.h>

// ---------------------------------------------------------------------------
// DecoderRNN for MI455X (gfx1250):
//   - small GEMMs + recurrence: f32 WMMA (V_WMMA_F32_16X16X4_F32), exact;
//     recurrence uses 4 independent accumulator chains (latency-bound path)
//     and GLOBAL_LOAD_ASYNC_TO_LDS_B128 (ASYNCcnt) for h_{t-1} staging.
//   - dominant logits GEMM: bf16 hi/lo 3-term split on
//     V_WMMA_F32_16X16X32_BF16 (~1e-5 rel err, ~2.7x fewer matrix cycles).
// ---------------------------------------------------------------------------
typedef __attribute__((ext_vector_type(2)))  float  v2f;
typedef __attribute__((ext_vector_type(8)))  float  v8f;
typedef __attribute__((ext_vector_type(4)))  __bf16 v4bf;
typedef __attribute__((ext_vector_type(8)))  __bf16 v8bf;
typedef __attribute__((ext_vector_type(16))) __bf16 v16bf;

#define B_SZ 64
#define SEQ  64
#define T_SZ 63
#define IMGF 2048
#define EMB  512
#define HID  512
#define VOC  32000
#define M_TOT (B_SZ * T_SZ)   // 4032

// f32-WMMA tiled-GEMM config: 64x128 tile, 256 threads = 8 waves
#define BM 64
#define BN 128
#define BK 16
#define ASTR 20    // LDS A stride (floats): [m][k]
#define BSTR 18    // LDS B^T stride (floats): [n][k]

// bf16 split-GEMM config (k_out): K-chunk 32
#define BK2 32
#define AS2 40     // bf16 elements per A row  [m][k], 16B-aligned rows
#define BS2 40     // bf16 elements per B^T row [n][k]

// RNN persistent-kernel LDS strides
#define HSTR 516
#define WSTR 516
#define NWG_RNN 16

__device__ __forceinline__ v8f zero_v8() {
  v8f v;
#pragma unroll
  for (int i = 0; i < 8; ++i) v[i] = 0.0f;
  return v;
}

__device__ __forceinline__ v8f wmma_f32(v2f a, v2f b, v8f c) {
  return __builtin_amdgcn_wmma_f32_16x16x4_f32(false, a, false, b, (short)0, c,
                                               false, false);
}

__device__ __forceinline__ v8f wmma_bf16(v16bf a, v16bf b, v8f c) {
  return __builtin_amdgcn_wmma_f32_16x16x32_bf16(false, a, false, b, (short)0, c,
                                                 false, false);
}

// combine two 8-element (16B, 16B-aligned) LDS chunks into one v16bf operand
__device__ __forceinline__ v16bf comb16(const __bf16* p, int o0, int o1) {
  const v8bf x = *(const v8bf*)(p + o0);
  const v8bf y = *(const v8bf*)(p + o1);
  v16bf r;
#pragma unroll
  for (int i = 0; i < 8; ++i) { r[i] = x[i]; r[i + 8] = y[i]; }
  return r;
}

// wave computes 16(M) x 64(N) strip of the 64x128 block over one BK=16 chunk
__device__ __forceinline__ void mma_chunk(const float* __restrict__ As,
                                          const float* __restrict__ Bs,
                                          int wm, int wn, int m, int hi,
                                          v8f acc[4]) {
#pragma unroll
  for (int kk = 0; kk < BK; kk += 4) {
    v2f a = *(const v2f*)&As[(wm * 16 + m) * ASTR + kk + 2 * hi];
#pragma unroll
    for (int j = 0; j < 4; ++j) {
      v2f b = *(const v2f*)&Bs[(wn * 64 + j * 16 + m) * BSTR + kk + 2 * hi];
      acc[j] = wmma_f32(a, b, acc[j]);
    }
  }
}

// ---------------------------------------------------------------------------
// Kernel 1: pre[b,t,:] = embed_W[captions[b,t]] @ Wx + bx  (M=4032,K=512,N=512)
// ---------------------------------------------------------------------------
__global__ __launch_bounds__(256) void k_pre(const int* __restrict__ cap,
                                             const float* __restrict__ embW,
                                             const float* __restrict__ Wx,
                                             const float* __restrict__ bx,
                                             float* __restrict__ preh) {
  __shared__ float As[BM * ASTR];
  __shared__ float Bs[BN * BSTR];
  const int tid = threadIdx.x;
  const int m0 = blockIdx.y * BM;
  const int n0 = blockIdx.x * BN;
  const int lane = tid & 31, wave = tid >> 5;
  const int wm = wave & 3, wn = wave >> 2;
  const int m = lane & 15, hi = lane >> 4;

  const int ar = tid >> 2;
  const int ac = (tid & 3) * 4;
  const int rr = m0 + ar;
  const int bb = rr / T_SZ;
  const int tt = rr - bb * T_SZ;
  const float* arow = embW + (size_t)cap[bb * SEQ + tt] * EMB;
  const int br = tid >> 5;
  const int bc = (tid & 31) * 4;

  v8f acc[4];
#pragma unroll
  for (int j = 0; j < 4; ++j) acc[j] = zero_v8();

  for (int k0 = 0; k0 < EMB; k0 += BK) {
    const float4 av = *(const float4*)(arow + k0 + ac);
    *(float4*)&As[ar * ASTR + ac] = av;
#pragma unroll
    for (int h = 0; h < 2; ++h) {
      const int kr = br + h * 8;
      const float4 bv = *(const float4*)(Wx + (size_t)(k0 + kr) * HID + n0 + bc);
      Bs[(bc + 0) * BSTR + kr] = bv.x;
      Bs[(bc + 1) * BSTR + kr] = bv.y;
      Bs[(bc + 2) * BSTR + kr] = bv.z;
      Bs[(bc + 3) * BSTR + kr] = bv.w;
    }
    __syncthreads();
    mma_chunk(As, Bs, wm, wn, m, hi, acc);
    __syncthreads();
  }
#pragma unroll
  for (int j = 0; j < 4; ++j) {
    const int col = n0 + wn * 64 + j * 16 + m;
    const float bxv = bx[col];
#pragma unroll
    for (int r = 0; r < 8; ++r) {
      const int row = m0 + wm * 16 + r + 8 * hi;
      preh[(size_t)row * HID + col] = acc[j][r] + bxv;
    }
  }
}

// ---------------------------------------------------------------------------
// Kernel 2: pre[b,0,:] += img @ Wi + bi  (M=64,K=2048,N=512); zeroes RNN cnt
// ---------------------------------------------------------------------------
__global__ __launch_bounds__(256) void k_img(const float* __restrict__ img,
                                             const float* __restrict__ Wi,
                                             const float* __restrict__ bi,
                                             float* __restrict__ preh,
                                             unsigned* __restrict__ cnt) {
  __shared__ float As[BM * ASTR];
  __shared__ float Bs[BN * BSTR];
  const int tid = threadIdx.x;
  const int n0 = blockIdx.x * BN;
  if (blockIdx.x == 0 && tid < 64) cnt[tid] = 0u;
  const int lane = tid & 31, wave = tid >> 5;
  const int wm = wave & 3, wn = wave >> 2;
  const int m = lane & 15, hi = lane >> 4;

  const int ar = tid >> 2;
  const int ac = (tid & 3) * 4;
  const float* arow = img + (size_t)ar * IMGF;
  const int br = tid >> 5;
  const int bc = (tid & 31) * 4;

  v8f acc[4];
#pragma unroll
  for (int j = 0; j < 4; ++j) acc[j] = zero_v8();

  for (int k0 = 0; k0 < IMGF; k0 += BK) {
    const float4 av = *(const float4*)(arow + k0 + ac);
    *(float4*)&As[ar * ASTR + ac] = av;
#pragma unroll
    for (int h = 0; h < 2; ++h) {
      const int kr = br + h * 8;
      const float4 bv = *(const float4*)(Wi + (size_t)(k0 + kr) * HID + n0 + bc);
      Bs[(bc + 0) * BSTR + kr] = bv.x;
      Bs[(bc + 1) * BSTR + kr] = bv.y;
      Bs[(bc + 2) * BSTR + kr] = bv.z;
      Bs[(bc + 3) * BSTR + kr] = bv.w;
    }
    __syncthreads();
    mma_chunk(As, Bs, wm, wn, m, hi, acc);
    __syncthreads();
  }
#pragma unroll
  for (int j = 0; j < 4; ++j) {
    const int col = n0 + wn * 64 + j * 16 + m;
    const float biv = bi[col];
#pragma unroll
    for (int r = 0; r < 8; ++r) {
      const int b = wm * 16 + r + 8 * hi;
      const size_t idx = (size_t)(b * T_SZ) * HID + col;
      preh[idx] += acc[j][r] + biv;
    }
  }
}

// ---------------------------------------------------------------------------
// Kernel 3: persistent RNN, 16 WGs, Wh^T slice + h_{t-1} in LDS (~198 KB/WGP).
// h_{t-1} staged with GLOBAL_LOAD_ASYNC_TO_LDS_B128 (ASYNCcnt), recurrence
// GEMM uses 4 independent WMMA accumulator chains to hide WMMA RAW latency.
// ---------------------------------------------------------------------------
__global__ __launch_bounds__(256) void k_rnn(const float* __restrict__ Wh,
                                             const float* __restrict__ bh,
                                             float* __restrict__ preh,
                                             float* __restrict__ hbuf,
                                             unsigned* __restrict__ cnt) {
  extern __shared__ float smem[];
  float* WhT = smem;               // [32][WSTR]
  float* Hs = smem + 32 * WSTR;    // [64][HSTR]
  const int tid = threadIdx.x;
  const int n0 = blockIdx.x * 32;
  const int lane = tid & 31, wave = tid >> 5;
  const int wm = wave & 3, wn = wave >> 2;
  const int m = lane & 15, hi = lane >> 4;

  {
    const int c = tid & 31;
    const int kb = (tid >> 5) * 64;
    for (int i = 0; i < 64; ++i) {
      const int k = kb + i;
      WhT[c * WSTR + k] = Wh[(size_t)k * HID + n0 + c];
    }
  }

#pragma unroll
  for (int i = 0; i < 8; ++i) {
    const int flat = tid + i * 256;
    const int b = flat >> 5, c = flat & 31;
    const int ng = n0 + c;
    const size_t idx = (size_t)(b * T_SZ) * HID + ng;
    const float h = tanhf(preh[idx] + bh[ng]);
    preh[idx] = h;
    hbuf[b * HID + ng] = h;
  }
  __threadfence();
  __syncthreads();
  if (tid == 0)
    __hip_atomic_fetch_add(&cnt[0], 1u, __ATOMIC_RELEASE, __HIP_MEMORY_SCOPE_AGENT);

  // dynamic-LDS byte offset of Hs (no static LDS in this kernel)
  const unsigned hs_lds_base = (unsigned)(32 * WSTR * sizeof(float));

  for (int t = 1; t < T_SZ; ++t) {
    const float* hr = hbuf + ((t - 1) & 1) * (B_SZ * HID);
    float* hw = hbuf + (t & 1) * (B_SZ * HID);
    if (tid == 0) {
      while (__hip_atomic_load(&cnt[t - 1], __ATOMIC_ACQUIRE,
                               __HIP_MEMORY_SCOPE_AGENT) < NWG_RNN)
        __builtin_amdgcn_s_sleep(1);
    }
    __syncthreads();
    // stage h_{t-1} (64x512 f32) straight into LDS: async DMA, no VGPR bounce
    for (int i = 0; i < 32; ++i) {
      const int flat = (tid + i * 256) * 4;
      const int b = flat >> 9;
      const int k = flat & 511;
      const unsigned dst = hs_lds_base + (unsigned)(b * HSTR + k) * 4u;
      const unsigned long long src =
          (unsigned long long)(uintptr_t)(hr + b * HID + k);
      asm volatile("global_load_async_to_lds_b128 %0, %1, off"
                   :: "v"(dst), "v"(src) : "memory");
    }
    asm volatile("s_wait_asynccnt 0x0" ::: "memory");
    __syncthreads();

    // 16x16 tile per wave, K=512: 4 interleaved accumulator chains
    v8f a0 = zero_v8(), a1 = zero_v8(), a2 = zero_v8(), a3 = zero_v8();
    const float* hrow = &Hs[(wm * 16 + m) * HSTR + 2 * hi];
    const float* wrow = &WhT[(wn * 16 + m) * WSTR + 2 * hi];
    for (int k = 0; k < HID; k += 16) {
      a0 = wmma_f32(*(const v2f*)(hrow + k),      *(const v2f*)(wrow + k),      a0);
      a1 = wmma_f32(*(const v2f*)(hrow + k + 4),  *(const v2f*)(wrow + k + 4),  a1);
      a2 = wmma_f32(*(const v2f*)(hrow + k + 8),  *(const v2f*)(wrow + k + 8),  a2);
      a3 = wmma_f32(*(const v2f*)(hrow + k + 12), *(const v2f*)(wrow + k + 12), a3);
    }
    v8f acc;
#pragma unroll
    for (int i = 0; i < 8; ++i) acc[i] = (a0[i] + a1[i]) + (a2[i] + a3[i]);

    const int ng = n0 + wn * 16 + m;
    const float bhv = bh[ng];
#pragma unroll
    for (int r = 0; r < 8; ++r) {
      const int b = wm * 16 + r + 8 * hi;
      const size_t idx = (size_t)(b * T_SZ + t) * HID + ng;
      const float h = tanhf(acc[r] + preh[idx] + bhv);
      preh[idx] = h;
      hw[b * HID + ng] = h;
    }
    __threadfence();
    __syncthreads();
    if (tid == 0)
      __hip_atomic_fetch_add(&cnt[t], 1u, __ATOMIC_RELEASE,
                             __HIP_MEMORY_SCOPE_AGENT);
  }
}

// ---------------------------------------------------------------------------
// Kernel 4: logits = hs @ Wy + by  (M=4032,K=512,N=32000)
// bf16 hi/lo 3-term split on v_wmma_f32_16x16x32_bf16; on-the-fly conversion
// while staging tiles to LDS; register prefetch pipeline; NT output stores.
// ---------------------------------------------------------------------------
__global__ __launch_bounds__(256) void k_out(const float* __restrict__ hs,
                                             const float* __restrict__ Wy,
                                             const float* __restrict__ by,
                                             float* __restrict__ out) {
  __shared__ __bf16 Ah[BM * AS2];
  __shared__ __bf16 Al[BM * AS2];
  __shared__ __bf16 Bh[BN * BS2];
  __shared__ __bf16 Bl[BN * BS2];
  const int tid = threadIdx.x;
  const int m0 = blockIdx.y * BM;
  const int n0 = blockIdx.x * BN;
  const int lane = tid & 31, wave = tid >> 5;
  const int wm = wave & 3, wn = wave >> 2;
  const int m = lane & 15, hi = lane >> 4;

  const int ar = tid >> 2;
  const int ac = (tid & 3) * 8;
  const float* arow = hs + (size_t)(m0 + ar) * HID;
  const int bn = (tid & 31) * 4;
  const int bk = (tid >> 5) * 4;

  v8f acc[4];
#pragma unroll
  for (int j = 0; j < 4; ++j) acc[j] = zero_v8();

  float4 a0 = *(const float4*)(arow + ac);
  float4 a1 = *(const float4*)(arow + ac + 4);
  float4 bv[4];
#pragma unroll
  for (int r = 0; r < 4; ++r)
    bv[r] = *(const float4*)(Wy + (size_t)(bk + r) * VOC + n0 + bn);

  for (int c = 0; c < HID / BK2; ++c) {
    {
      float av[8];
      *(float4*)&av[0] = a0;
      *(float4*)&av[4] = a1;
      v8bf h8, l8;
#pragma unroll
      for (int i = 0; i < 8; ++i) {
        const __bf16 h = (__bf16)av[i];
        h8[i] = h;
        l8[i] = (__bf16)(av[i] - (float)h);
      }
      *(v8bf*)&Ah[ar * AS2 + ac] = h8;
      *(v8bf*)&Al[ar * AS2 + ac] = l8;
    }
#pragma unroll
    for (int j = 0; j < 4; ++j) {
      v4bf h4, l4;
#pragma unroll
      for (int r = 0; r < 4; ++r) {
        const float x = ((const float*)&bv[r])[j];
        const __bf16 h = (__bf16)x;
        h4[r] = h;
        l4[r] = (__bf16)(x - (float)h);
      }
      *(v4bf*)&Bh[(bn + j) * BS2 + bk] = h4;
      *(v4bf*)&Bl[(bn + j) * BS2 + bk] = l4;
    }
    __syncthreads();

    if (c + 1 < HID / BK2) {
      const int kn = (c + 1) * BK2;
      a0 = *(const float4*)(arow + kn + ac);
      a1 = *(const float4*)(arow + kn + ac + 4);
#pragma unroll
      for (int r = 0; r < 4; ++r)
        bv[r] = *(const float4*)(Wy + (size_t)(kn + bk + r) * VOC + n0 + bn);
    }

    const __bf16* arA = &Ah[(wm * 16 + m) * AS2];
    const __bf16* arAl = &Al[(wm * 16 + m) * AS2];
    const v16bf ah = comb16(arA, 8 * hi, 16 + 8 * hi);
    const v16bf al = comb16(arAl, 8 * hi, 16 + 8 * hi);
#pragma unroll
    for (int j = 0; j < 4; ++j) {
      const int nrow = (wn * 64 + j * 16 + m) * BS2;
      const v16bf bh16 = comb16(&Bh[nrow], 8 * hi, 16 + 8 * hi);
      const v16bf bl16 = comb16(&Bl[nrow], 8 * hi, 16 + 8 * hi);
      acc[j] = wmma_bf16(ah, bh16, acc[j]);
      acc[j] = wmma_bf16(ah, bl16, acc[j]);
      acc[j] = wmma_bf16(al, bh16, acc[j]);
    }
    __syncthreads();
  }

#pragma unroll
  for (int j = 0; j < 4; ++j) {
    const int col = n0 + wn * 64 + j * 16 + m;
    const float byv = by[col];
#pragma unroll
    for (int r = 0; r < 8; ++r) {
      const int row = m0 + wm * 16 + r + 8 * hi;
      __builtin_nontemporal_store(acc[j][r] + byv, &out[(size_t)row * VOC + col]);
    }
  }
}

// ---------------------------------------------------------------------------
extern "C" void kernel_launch(void* const* d_in, const int* in_sizes, int n_in,
                              void* d_out, int out_size, void* d_ws,
                              size_t ws_size, hipStream_t stream) {
  const float* img = (const float*)d_in[0];
  const int* cap = (const int*)d_in[1];
  const float* embW = (const float*)d_in[2];
  const float* Wi = (const float*)d_in[3];
  const float* bi = (const float*)d_in[4];
  const float* Wx = (const float*)d_in[5];
  const float* bx = (const float*)d_in[6];
  const float* Wh = (const float*)d_in[7];
  const float* bh = (const float*)d_in[8];
  const float* Wy = (const float*)d_in[9];
  const float* by = (const float*)d_in[10];
  float* out = (float*)d_out;

  float* preh = (float*)d_ws;                          // 4032*512 f32 (pre -> hs)
  float* hbuf = preh + (size_t)M_TOT * HID;            // 2 * 64*512 f32 ping-pong
  unsigned* cnt = (unsigned*)(hbuf + 2 * B_SZ * HID);  // 64 step counters

  dim3 g1(HID / BN, M_TOT / BM);  // (4, 63)
  k_pre<<<g1, 256, 0, stream>>>(cap, embW, Wx, bx, preh);

  dim3 g2(HID / BN, 1);  // (4, 1)
  k_img<<<g2, 256, 0, stream>>>(img, Wi, bi, preh, cnt);

  const size_t rnn_lds = (size_t)(32 * WSTR + 64 * HSTR) * sizeof(float);  // ~198 KB
  k_rnn<<<NWG_RNN, 256, rnn_lds, stream>>>(Wh, bh, preh, hbuf, cnt);

  dim3 g4(VOC / BN, M_TOT / BM);  // (250, 63)
  k_out<<<g4, 256, 0, stream>>>(preh, Wy, by, out);
}
